// GCN_77214922048113
// MI455X (gfx1250) — matrix operand order repaired and checked by reading, not compiled, not run on hardware
//
#include <hip/hip_runtime.h>

// GCN forward: out = relu(adj @ (x @ W) + b)
//   x:[32,1024,128] f32, adj:[32,1024,1024] f32, W:[128,128] f32, b:[128] f32
// Strategy: bf16 WMMA (v_wmma_f32_16x16x32_bf16) with f32 accumulation.
// HBM-bound on streaming adj (134 MB) exactly once; support kept as bf16
// transposed [b][h][m] in workspace so B-operand lane loads are contiguous.

#define NFEAT  128
#define NHID   128
#define NNODE  1024
#define NBATCH 32

typedef __attribute__((ext_vector_type(16))) __bf16 v16bf;
typedef __attribute__((ext_vector_type(8)))  __bf16 v8bf;
typedef __attribute__((ext_vector_type(8)))  float  v8f;
typedef __attribute__((ext_vector_type(4)))  float  v4f;

// Pack 16 f32 (as four v4f, in K order) into a v16bf WMMA operand.
__device__ __forceinline__ v16bf pack_bf16(v4f f0, v4f f1, v4f f2, v4f f3) {
  v16bf r;
#pragma unroll
  for (int i = 0; i < 4; ++i) {
    r[i]      = (__bf16)f0[i];
    r[4 + i]  = (__bf16)f1[i];
    r[8 + i]  = (__bf16)f2[i];
    r[12 + i] = (__bf16)f3[i];
  }
  return r;
}

// ---------------- Kernel 0: W[f][h] f32 -> WT[h][f] bf16 (32 KB) -------------
__global__ void wt_prep(const float* __restrict__ W, __bf16* __restrict__ WT) {
  const int f = threadIdx.x;   // 0..127
  const int h = blockIdx.x;    // 0..127
  WT[h * NFEAT + f] = (__bf16)W[f * NHID + h];
}

// ------------- Kernel A: supportT[b][h][m] = sum_f WT[h][f] * x[b][m][f] -----
// GEMM: M'=128 (hid), N'=32768 (flat node), K=128 (feat).
// Block = 8 waves; wave w -> hid rows 16w..16w+15, all 128 node columns.
__global__ void __launch_bounds__(256)
support_gemm(const float* __restrict__ X,    // [32768][128] f32 (x flattened)
             const __bf16* __restrict__ WT,  // [128][128] bf16
             __bf16* __restrict__ ST)        // [32][128][1024] bf16
{
  const int lane = threadIdx.x & 31;
  const int wave = threadIdx.x >> 5;
  const int l16  = lane & 15;
  const int half = lane >> 4;

  const int colBase = blockIdx.x * 128;      // flat node base (block stays in one batch)
  const int batch   = colBase >> 10;
  const int mBase   = colBase & 1023;        // node base within batch
  const int hRow    = wave * 16 + l16;       // A-fragment row (hid index)

  v8f acc[8];
  const v8f z = {0.f,0.f,0.f,0.f,0.f,0.f,0.f,0.f};
#pragma unroll
  for (int t = 0; t < 8; ++t) acc[t] = z;

#pragma unroll
  for (int ks = 0; ks < NFEAT; ks += 32) {
    // A fragment: WT row, K = ks + half*16 .. +15 contiguous
    const __bf16* ap = WT + hRow * NFEAT + ks + half * 16;
    v16bf a;
    {
      v8bf c0 = *(const v8bf*)(ap);
      v8bf c1 = *(const v8bf*)(ap + 8);
#pragma unroll
      for (int i = 0; i < 8; ++i) { a[i] = c0[i]; a[8 + i] = c1[i]; }
    }
#pragma unroll
    for (int t = 0; t < 8; ++t) {
      // B fragment: lane column = node row of x; K = ks + half*16 .. +15 contiguous
      const int nRow = colBase + t * 16 + l16;
      const v4f* q = (const v4f*)(X + (size_t)nRow * NFEAT + ks + half * 16);
      v16bf b = pack_bf16(q[0], q[1], q[2], q[3]);
      acc[t] = __builtin_amdgcn_wmma_f32_16x16x32_bf16(
          false, a, false, b, (short)0, acc[t], false, false);
    }
  }

  // C layout: VGPR v -> M = v + 8*half (hid); lane&15 -> N (node). Coalesced bf16 stores.
#pragma unroll
  for (int t = 0; t < 8; ++t) {
#pragma unroll
    for (int v = 0; v < 8; ++v) {
      const int h = wave * 16 + v + half * 8;
      const int m = mBase + t * 16 + l16;
      ST[((size_t)batch * NHID + h) * NNODE + m] = (__bf16)acc[t][v];
    }
  }
}

// ------------- Kernel B: out = relu(adj @ support + bias) --------------------
// Per batch GEMM: M=1024 (node), N=128 (hid), K=1024 (node).
// Grid (8, 32): blockIdx.x = M tile of 128, blockIdx.y = batch.
__global__ void __launch_bounds__(256)
gcn_main(const float* __restrict__ ADJ,   // [32][1024][1024] f32
         const __bf16* __restrict__ ST,   // [32][128][1024] bf16 (supportT)
         const float* __restrict__ BIAS,  // [128] f32
         float* __restrict__ OUT)         // [32][1024][128] f32
{
  const int lane = threadIdx.x & 31;
  const int wave = threadIdx.x >> 5;
  const int l16  = lane & 15;
  const int half = lane >> 4;

  const int batch = blockIdx.y;
  const int mTile = blockIdx.x * 128 + wave * 16;  // wave's 16-row strip
  const int aRow  = mTile + l16;                   // A-fragment row (adj row)

  const float* adjRow = ADJ + ((size_t)batch * NNODE + aRow) * NNODE;
  const __bf16* stB   = ST + (size_t)batch * NHID * NNODE;

  v8f acc[8];
  const v8f z = {0.f,0.f,0.f,0.f,0.f,0.f,0.f,0.f};
#pragma unroll
  for (int t = 0; t < 8; ++t) acc[t] = z;

  for (int ks = 0; ks < NNODE; ks += 32) {
    // A fragment: adj row, f32 -> bf16. K = ks+half*16 .. +15 contiguous.
    const v4f* p0 = (const v4f*)(adjRow + ks + half * 16);
    const v4f* p1 = (const v4f*)(adjRow + ks + half * 16 + 8);
    v4f f0 = __builtin_nontemporal_load(p0);      // adj is streamed once: NT hint
    v4f f1 = __builtin_nontemporal_load(p0 + 1);
    v4f f2 = __builtin_nontemporal_load(p1);
    v4f f3 = __builtin_nontemporal_load(p1 + 1);
    v16bf a = pack_bf16(f0, f1, f2, f3);

#pragma unroll
    for (int t = 0; t < 8; ++t) {
      // B fragment: lane column h = t*16 + l16; K contiguous in supportT.
      const __bf16* bp = stB + ((size_t)(t * 16 + l16)) * NNODE + ks + half * 16;
      v16bf b = *(const v16bf*)bp;
      acc[t] = __builtin_amdgcn_wmma_f32_16x16x32_bf16(
          false, a, false, b, (short)0, acc[t], false, false);
    }
  }

  // Epilogue: bias + relu. C layout: VGPR v -> row mTile + v + 8*half; lane -> hid col.
#pragma unroll
  for (int t = 0; t < 8; ++t) {
    const int h = t * 16 + l16;
    const float bias = BIAS[h];
#pragma unroll
    for (int v = 0; v < 8; ++v) {
      const int r = mTile + v + half * 8;
      const float val = acc[t][v] + bias;
      OUT[((size_t)batch * NNODE + r) * NHID + h] = fmaxf(val, 0.0f);
    }
  }
}

// ----------------------------------------------------------------------------
extern "C" void kernel_launch(void* const* d_in, const int* in_sizes, int n_in,
                              void* d_out, int out_size, void* d_ws, size_t ws_size,
                              hipStream_t stream) {
  const float* x    = (const float*)d_in[0];  // [32,1024,128]
  const float* adj  = (const float*)d_in[1];  // [32,1024,1024]
  const float* W    = (const float*)d_in[2];  // [128,128]
  const float* bias = (const float*)d_in[3];  // [128]
  float* out = (float*)d_out;                 // [32,1024,128]

  // Workspace: supportT bf16 (8 MB) then WT bf16 (32 KB). Needs ~8.4 MB of ws.
  __bf16* ST = (__bf16*)d_ws;
  __bf16* WT = (__bf16*)((char*)d_ws +
                         (size_t)NBATCH * NHID * NNODE * sizeof(__bf16));

  wt_prep<<<dim3(NHID), dim3(NFEAT), 0, stream>>>(W, WT);
  support_gemm<<<dim3((NBATCH * NNODE) / 128), dim3(256), 0, stream>>>(x, WT, ST);
  gcn_main<<<dim3(NNODE / 128, NBATCH), dim3(256), 0, stream>>>(adj, ST, bias, out);
}
